// WanSelfAttention_masked_KV_55980603736348
// MI455X (gfx1250) — compile-verified, matrix operand order checked by reading, not run on hardware
//
#include <hip/hip_runtime.h>
#include <hip/hip_bf16.h>

#define DIM_ 1536
#define NH_  12
#define HD_  128
#define S_   2048

typedef __attribute__((ext_vector_type(16))) __bf16    v16bf;
typedef __attribute__((ext_vector_type(8)))  __bf16    v8bf;
typedef __attribute__((ext_vector_type(8)))  float     v8f;
typedef __attribute__((ext_vector_type(4)))  unsigned  v4u;
typedef __attribute__((ext_vector_type(4)))  int       v4i;
typedef __attribute__((ext_vector_type(8)))  int       v8i;

static __device__ __forceinline__ unsigned short f2bf_bits(float f) {
  union { float f; unsigned u; } v; v.f = f;
  unsigned r = v.u + 0x7FFFu + ((v.u >> 16) & 1u);
  return (unsigned short)(r >> 16);
}

// Build a 16x32 bf16 A/B fragment for v_wmma_f32_16x16x32_bf16.
// Per ISA 7.12.2: lane holds two contiguous 8-element runs: cols [8h..8h+7]
// and [16+8h..16+8h+7] of its row => two b128 loads.
static __device__ __forceinline__ v16bf frag_from(const unsigned char* p0,
                                                  const unsigned char* p1) {
  v16bf r;
  ((v8bf*)&r)[0] = *(const v8bf*)p0;
  ((v8bf*)&r)[1] = *(const v8bf*)p1;
  return r;
}

// CDNA5 async copy: global -> LDS, 16 bytes per lane, tracked by ASYNCcnt.
static __device__ __forceinline__ void async_g2l_b128(unsigned lds_addr,
                                                      const void* gaddr) {
  asm volatile("global_load_async_to_lds_b128 %0, %1, off"
               :: "v"(lds_addr), "v"(gaddr)
               : "memory");
}
static __device__ __forceinline__ void wait_asynccnt0() {
  asm volatile("s_wait_asynccnt 0x0" ::: "memory");
}

// ---------------------------------------------------------------------------
// Tensor Data Mover: DMA a 2D bf16 tile (tile_d0 contiguous elems x tile_d1
// rows, row stride = stride_elems) from global memory into LDS (packed
// row-major, tile_d0*2 bytes per row). D# per ISA ch.8; tracked by TENSORcnt.
// ---------------------------------------------------------------------------
static __device__ __forceinline__ void tdm_load_2d_bf16(unsigned lds_addr,
                                                        const void* gaddr,
                                                        unsigned tile_d0,
                                                        unsigned tile_d1,
                                                        unsigned stride_elems) {
  unsigned long long ga = (unsigned long long)gaddr;
  v4u g0;
  g0[0] = 1u;                                              // count=1, user D#
  g0[1] = lds_addr;                                        // lds_addr
  g0[2] = (unsigned)(ga & 0xFFFFFFFFu);                    // global_addr[31:0]
  g0[3] = (unsigned)((ga >> 32) & 0x01FFFFFFu) | 0x80000000u; // [56:32] | type=2

  v8i g1;
  g1[0] = 0x10000;                                         // data_size=1 (2B)
  g1[1] = (int)((tile_d0 & 0xFFFFu) << 16);                // tensor_dim0[15:0]
  g1[2] = (int)((tile_d0 >> 16) | ((tile_d1 & 0xFFFFu) << 16)); // d0 hi|d1 lo
  g1[3] = (int)((tile_d1 >> 16) | (tile_d0 << 16));        // d1 hi | tile_dim0
  g1[4] = (int)(tile_d1 & 0xFFFFu);                        // tile_dim1 (dim2=0)
  g1[5] = (int)stride_elems;                               // dim0_stride[31:0]
  g1[6] = 0;                                               // stride hi, d1s lo
  g1[7] = 0;
  v4i z4 = {0, 0, 0, 0};
#if defined(__clang_major__) && (__clang_major__ >= 23)
  v8i z8 = {0, 0, 0, 0, 0, 0, 0, 0};
  __builtin_amdgcn_tensor_load_to_lds(g0, g1, z4, z4, z8, 0);
#else
  __builtin_amdgcn_tensor_load_to_lds(g0, g1, z4, z4, 0);
#endif
}

// ---------------------------------------------------------------------------
// fp32 -> bf16 elementwise convert
// ---------------------------------------------------------------------------
__global__ __launch_bounds__(256)
void f32_to_bf16_kernel(const float* __restrict__ in,
                        unsigned short* __restrict__ out, int n) {
  int i = blockIdx.x * blockDim.x + threadIdx.x;
  if (i < n) out[i] = f2bf_bits(in[i]);
}

// ---------------------------------------------------------------------------
// C[M,N] = A[M,K](bf16, row-major) * B[N,K]^T(bf16, row-major) + bias[N]
// mode 0: fp32 row-major out; mode 1: bf16 row-major; mode 2: bf16 transposed
// Block: 256 thr (8 waves), 64x64 tile; wave = 16(M) x 32(N); k-step 32.
// Double-buffered GLOBAL_LOAD_ASYNC_TO_LDS_B128 staging.
// ---------------------------------------------------------------------------
__global__ __launch_bounds__(256)
void gemm_bf16_wmma(const unsigned short* __restrict__ A,
                    const unsigned short* __restrict__ B,
                    const float* __restrict__ bias,
                    void* __restrict__ Cout,
                    int M, int N, int K, int mode) {
  __shared__ __align__(16) unsigned char sA[2][64 * 64]; // 64 rows x 32 bf16
  __shared__ __align__(16) unsigned char sB[2][64 * 64];

  const int tid  = threadIdx.x;
  const int wave = tid >> 5;
  const int lane = tid & 31;
  const int half = lane >> 4;
  const int l16  = lane & 15;
  const int wy   = wave >> 1;   // 0..3 -> M sub-tile
  const int wx   = wave & 1;    // 0..1 -> N sub-tile (32 wide)

  const int mBase = blockIdx.y * 64;
  const int nBase = blockIdx.x * 64;

  const int lrow = tid >> 2;        // 0..63
  const int lcol = (tid & 3) * 16;  // byte offset within 64B row

  const unsigned char* Ab = (const unsigned char*)A;
  const unsigned char* Bb = (const unsigned char*)B;
  const long aRow = (long)(mBase + lrow) * K * 2 + lcol;
  const long bRow = (long)(nBase + lrow) * K * 2 + lcol;

  // Per-lane LDS destinations for the async engine (double buffered).
  unsigned stA[2], stB[2];
  stA[0] = (unsigned)(uintptr_t)(&sA[0][lrow * 64 + lcol]);
  stA[1] = (unsigned)(uintptr_t)(&sA[1][lrow * 64 + lcol]);
  stB[0] = (unsigned)(uintptr_t)(&sB[0][lrow * 64 + lcol]);
  stB[1] = (unsigned)(uintptr_t)(&sB[1][lrow * 64 + lcol]);

  v8f acc0 = {};
  v8f acc1 = {};

  // Prologue: stream tile 0 into buffer 0.
  async_g2l_b128(stA[0], Ab + aRow);
  async_g2l_b128(stB[0], Bb + bRow);
  wait_asynccnt0();
  __syncthreads();

  int cur = 0;
  for (int k0 = 0; k0 < K; k0 += 32) {
    const int nxt = cur ^ 1;
    if (k0 + 32 < K) {
      // Stream next tile while we compute on the current one.
      async_g2l_b128(stA[nxt], Ab + aRow + (long)(k0 + 32) * 2);
      async_g2l_b128(stB[nxt], Bb + bRow + (long)(k0 + 32) * 2);
      // Distance-2 speculative prefetch keeps L2 ahead of the async engine.
      __builtin_prefetch(Ab + aRow + (long)(k0 + 64) * 2, 0, 1);
      __builtin_prefetch(Bb + bRow + (long)(k0 + 64) * 2, 0, 1);
    }

    const unsigned char* ar = sA[cur] + (wy * 16 + l16) * 64;
    const unsigned char* b0 = sB[cur] + (wx * 32 + l16) * 64;
    const unsigned char* b1 = sB[cur] + (wx * 32 + 16 + l16) * 64;
    v16bf af  = frag_from(ar + half * 16, ar + 32 + half * 16);
    v16bf bf0 = frag_from(b0 + half * 16, b0 + 32 + half * 16);
    v16bf bf1 = frag_from(b1 + half * 16, b1 + 32 + half * 16);

    acc0 = __builtin_amdgcn_wmma_f32_16x16x32_bf16(false, af, false, bf0,
                                                   (short)0, acc0, false, false);
    acc1 = __builtin_amdgcn_wmma_f32_16x16x32_bf16(false, af, false, bf1,
                                                   (short)0, acc1, false, false);

    wait_asynccnt0();   // own next-tile async writes landed in LDS
    __syncthreads();    // everyone's writes visible; cur fully consumed
    cur = nxt;
  }

  const int m0 = mBase + wy * 16;
  const int n0 = nBase + wx * 32;
  for (int c = 0; c < 2; ++c) {
    v8f acc = c ? acc1 : acc0;
    int n = n0 + c * 16 + l16;
    float bb = bias ? bias[n] : 0.0f;
    for (int r = 0; r < 8; ++r) {
      int m = m0 + r + half * 8;
      float v = acc[r] + bb;
      if (mode == 0)      ((float*)Cout)[(long)m * N + n] = v;
      else if (mode == 1) ((unsigned short*)Cout)[(long)m * N + n] = f2bf_bits(v);
      else                ((unsigned short*)Cout)[(long)n * M + m] = f2bf_bits(v);
    }
  }
}

// ---------------------------------------------------------------------------
// Fused RMSNorm(*g) + 3D RoPE, fp32 raw -> bf16 out. One block per row.
// ---------------------------------------------------------------------------
__global__ __launch_bounds__(256)
void rmsnorm_rope_kernel(const float* __restrict__ raw,
                         const float* __restrict__ g,
                         const float* __restrict__ freqs,
                         const int* __restrict__ grid_sizes,
                         unsigned short* __restrict__ out) {
  __shared__ float red[8];
  __shared__ float scale_sh;
  const int row = blockIdx.x;
  const int tid = threadIdx.x;
  const float* r = raw + (long)row * DIM_;

  float s = 0.0f;
  for (int i = tid; i < DIM_; i += 256) { float v = r[i]; s += v * v; }
  for (int off = 16; off; off >>= 1) s += __shfl_xor(s, off, 32);
  if ((tid & 31) == 0) red[tid >> 5] = s;
  __syncthreads();
  if (tid == 0) {
    float t = 0.0f;
    for (int i = 0; i < 8; ++i) t += red[i];
    scale_sh = rsqrtf(t / (float)DIM_ + 1e-6f);
  }
  __syncthreads();
  const float scale = scale_sh;

  const int hdim = grid_sizes[1], wdim = grid_sizes[2];
  const int fi = row / (hdim * wdim);
  const int hi = (row / wdim) % hdim;
  const int wi = row % wdim;
  const int c  = HD_ / 2;                 // 64
  const int s0 = c - 2 * (c / 3);         // 22
  const int s1 = c / 3;                   // 21

  for (int p = tid; p < NH_ * c; p += 256) {
    int nh = p / c, j = p % c;
    int sel = (j < s0) ? fi : ((j < s0 + s1) ? hi : wi);
    float ang = freqs[sel * c + j];
    float cs = __cosf(ang), sn = __sinf(ang);
    int base = nh * HD_ + 2 * j;
    float x0 = r[base]     * scale * g[base];
    float x1 = r[base + 1] * scale * g[base + 1];
    out[(long)row * DIM_ + base]     = f2bf_bits(x0 * cs - x1 * sn);
    out[(long)row * DIM_ + base + 1] = f2bf_bits(x0 * sn + x1 * cs);
  }
}

// ---------------------------------------------------------------------------
// Flash attention. Grid = NH * S/64 blocks, 128 threads (4 waves).
// All 4 waves of a block share one head and one 32-key block, so the K tile
// (32x128 bf16) and V tile (128x32 bf16, from V^T) are staged into LDS ONCE
// per block by the Tensor Data Mover (wave0 -> K, wave1 -> V), double
// buffered: TDM streams tile j+1 while WMMAs consume tile j. In-order
// per-wave TDM means s_wait_tensorcnt<=1 releases the current tile.
// ---------------------------------------------------------------------------
__global__ __launch_bounds__(128)
void attention_kernel(const unsigned short* __restrict__ qb,
                      const unsigned short* __restrict__ kb,
                      const unsigned short* __restrict__ vT,
                      const int* __restrict__ seq_lens,
                      unsigned short* __restrict__ attn) {
  __shared__ __align__(16) unsigned char sK[2][32 * 256];  // 32 keys x 128 bf16
  __shared__ __align__(16) unsigned char sV[2][128 * 64];  // 128 hd x 32 bf16
  __shared__ __align__(16) unsigned char pS[4 * 16 * 64];  // per-wave P tile

  const int wave = threadIdx.x >> 5;
  const int lane = threadIdx.x & 31;
  const int half = lane >> 4;
  const int l16  = lane & 15;

  const int head = blockIdx.x >> 5;     // 32 q-blocks of 64 per head
  const int qblk = blockIdx.x & 31;
  const int q0   = qblk * 64 + wave * 16;
  const int sl   = seq_lens[0];

  const unsigned char* Qb = (const unsigned char*)qb;
  const unsigned char* Kb = (const unsigned char*)kb;
  const unsigned char* Vb = (const unsigned char*)vT;
  unsigned char* myP = pS + wave * 1024;

  const unsigned ldsK[2] = { (unsigned)(uintptr_t)sK[0], (unsigned)(uintptr_t)sK[1] };
  const unsigned ldsV[2] = { (unsigned)(uintptr_t)sV[0], (unsigned)(uintptr_t)sV[1] };
  const unsigned char* kTile0 = Kb + head * (HD_ * 2);             // + j*DIM*2
  const unsigned char* vTile0 = Vb + (long)(head * HD_) * (S_ * 2); // + j*2

  // Q fragments: 16 rows x 128 hd = 4 chunks of K=32
  v16bf qf[4];
  {
    const long qrow = (long)(q0 + l16) * (DIM_ * 2) + head * (HD_ * 2);
    for (int kc = 0; kc < 4; ++kc)
      qf[kc] = frag_from(Qb + qrow + kc * 64 + half * 16,
                         Qb + qrow + kc * 64 + 32 + half * 16);
  }

  v8f o[8];
  for (int i = 0; i < 8; ++i) o[i] = (v8f){};
  float mrow[8], lrow[8];
  for (int r = 0; r < 8; ++r) { mrow[r] = -3.0e38f; lrow[r] = 0.0f; }

  const float scl = 0.08838834764831845f; // 1/sqrt(128)

  // TDM prologue: stream tile 0 into buffer 0.
  if (wave == 0) tdm_load_2d_bf16(ldsK[0], kTile0, HD_, 32, DIM_);
  if (wave == 1) tdm_load_2d_bf16(ldsV[0], vTile0, 32, HD_, S_);

  int cur = 0;
  for (int j0 = 0; j0 < S_; j0 += 32) {
    const int nxt = cur ^ 1;
    if (j0 + 32 < S_) {
      if (wave == 0)
        tdm_load_2d_bf16(ldsK[nxt], kTile0 + (long)(j0 + 32) * (DIM_ * 2), HD_, 32, DIM_);
      if (wave == 1)
        tdm_load_2d_bf16(ldsV[nxt], vTile0 + (long)(j0 + 32) * 2, 32, HD_, S_);
      if (wave < 2) __builtin_amdgcn_s_wait_tensorcnt(1); // cur tile landed
    } else {
      if (wave < 2) __builtin_amdgcn_s_wait_tensorcnt(0);
    }
    __syncthreads();  // cur K/V tiles visible to all waves

    const unsigned char* kT = sK[cur];
    const unsigned char* vTl = sV[cur];

    // ---- S = Q K^T for 32 keys (two 16x16 C tiles) ----
    v8f sa = {}, sb = {};
    for (int kc = 0; kc < 4; ++kc) {
      const unsigned char* k0p = kT + l16 * 256 + kc * 64;
      const unsigned char* k1p = kT + (16 + l16) * 256 + kc * 64;
      v16bf kf0 = frag_from(k0p + half * 16, k0p + 32 + half * 16);
      v16bf kf1 = frag_from(k1p + half * 16, k1p + 32 + half * 16);
      sa = __builtin_amdgcn_wmma_f32_16x16x32_bf16(false, qf[kc], false, kf0,
                                                   (short)0, sa, false, false);
      sb = __builtin_amdgcn_wmma_f32_16x16x32_bf16(false, qf[kc], false, kf1,
                                                   (short)0, sb, false, false);
    }

    // ---- online softmax over this 32-key block ----
    const int col0 = j0 + l16;
    const int col1 = j0 + 16 + l16;
    for (int r = 0; r < 8; ++r) {
      float a = sa[r] * scl; if (col0 >= sl) a = -3.0e38f;
      float b = sb[r] * scl; if (col1 >= sl) b = -3.0e38f;
      float cm = fmaxf(a, b);
      for (int off = 1; off < 16; off <<= 1) cm = fmaxf(cm, __shfl_xor(cm, off, 32));
      float mnew  = fmaxf(mrow[r], cm);
      float alpha = __expf(mrow[r] - mnew);
      float p0 = __expf(a - mnew);
      float p1 = __expf(b - mnew);
      float ps = p0 + p1;
      for (int off = 1; off < 16; off <<= 1) ps += __shfl_xor(ps, off, 32);
      lrow[r] = lrow[r] * alpha + ps;
      mrow[r] = mnew;
      for (int oc = 0; oc < 8; ++oc) o[oc][r] *= alpha;
      // stash P (bf16) in LDS, row = r + 8*half, cols l16 / 16+l16
      ((unsigned short*)myP)[(r + 8 * half) * 32 + l16]      = f2bf_bits(p0);
      ((unsigned short*)myP)[(r + 8 * half) * 32 + 16 + l16] = f2bf_bits(p1);
    }
    asm volatile("s_wait_dscnt 0" ::: "memory"); // per-wave LDS RAW fence

    // ---- O += P V : P is one 16x32 A-fragment, V tile gives contiguous B frags
    v16bf pf = frag_from(myP + l16 * 64 + half * 16,
                         myP + l16 * 64 + 32 + half * 16);
    for (int oc = 0; oc < 8; ++oc) {
      const unsigned char* vp = vTl + (oc * 16 + l16) * 64;
      v16bf vf = frag_from(vp + half * 16, vp + 32 + half * 16);
      o[oc] = __builtin_amdgcn_wmma_f32_16x16x32_bf16(false, pf, false, vf,
                                                      (short)0, o[oc], false, false);
    }

    __syncthreads();  // all reads of cur done before TDM may overwrite it
    cur = nxt;
  }

  // ---- normalize and store bf16 row-major ----
  for (int oc = 0; oc < 8; ++oc) {
    for (int r = 0; r < 8; ++r) {
      float v = o[oc][r] / lrow[r];
      int m = q0 + r + 8 * half;
      int n = head * HD_ + oc * 16 + l16;
      attn[(long)m * DIM_ + n] = f2bf_bits(v);
    }
  }
}

// ---------------------------------------------------------------------------
extern "C" void kernel_launch(void* const* d_in, const int* in_sizes, int n_in,
                              void* d_out, int out_size, void* d_ws, size_t ws_size,
                              hipStream_t stream) {
  (void)in_sizes; (void)n_in; (void)out_size; (void)ws_size;
  const float* x      = (const float*)d_in[0];
  const float* freqs  = (const float*)d_in[1];
  const float* wq     = (const float*)d_in[2];
  const float* bq     = (const float*)d_in[3];
  const float* wk     = (const float*)d_in[4];
  const float* bk     = (const float*)d_in[5];
  const float* wv     = (const float*)d_in[6];
  const float* bv     = (const float*)d_in[7];
  const float* wo     = (const float*)d_in[8];
  const float* bo     = (const float*)d_in[9];
  const float* gq     = (const float*)d_in[10];
  const float* gk     = (const float*)d_in[11];
  const int*   seqlen = (const int*)d_in[12];
  const int*   gsz    = (const int*)d_in[13];

  unsigned char* ws = (unsigned char*)d_ws;
  size_t off = 0;
  auto carve = [&](size_t bytes) -> unsigned char* {
    unsigned char* p = ws + off;
    off = (off + bytes + 255) & ~(size_t)255;
    return p;
  };

  const size_t SD2 = (size_t)S_ * DIM_ * 2;
  const size_t SD4 = (size_t)S_ * DIM_ * 4;
  const size_t DD2 = (size_t)DIM_ * DIM_ * 2;

  unsigned short* xb   = (unsigned short*)carve(SD2);
  unsigned short* wqb  = (unsigned short*)carve(DD2);
  unsigned short* wkb  = (unsigned short*)carve(DD2);
  unsigned short* wvb  = (unsigned short*)carve(DD2);
  unsigned short* wob  = (unsigned short*)carve(DD2);
  float*          qraw = (float*)carve(SD4);
  float*          kraw = (float*)carve(SD4);
  unsigned short* qbuf = (unsigned short*)carve(SD2);
  unsigned short* kbuf = (unsigned short*)carve(SD2);
  unsigned short* vTb  = (unsigned short*)carve(SD2);
  unsigned short* attn = (unsigned short*)carve(SD2);

  // 1) fp32 -> bf16 conversions
  f32_to_bf16_kernel<<<(S_ * DIM_ + 255) / 256, 256, 0, stream>>>(x, xb, S_ * DIM_);
  f32_to_bf16_kernel<<<(DIM_ * DIM_ + 255) / 256, 256, 0, stream>>>(wq, wqb, DIM_ * DIM_);
  f32_to_bf16_kernel<<<(DIM_ * DIM_ + 255) / 256, 256, 0, stream>>>(wk, wkb, DIM_ * DIM_);
  f32_to_bf16_kernel<<<(DIM_ * DIM_ + 255) / 256, 256, 0, stream>>>(wv, wvb, DIM_ * DIM_);
  f32_to_bf16_kernel<<<(DIM_ * DIM_ + 255) / 256, 256, 0, stream>>>(wo, wob, DIM_ * DIM_);

  // 2) QKV projections (WMMA GEMMs, async double-buffered staging)
  dim3 gg(DIM_ / 64, S_ / 64);
  gemm_bf16_wmma<<<gg, 256, 0, stream>>>(xb, wqb, bq, qraw, S_, DIM_, DIM_, 0);
  gemm_bf16_wmma<<<gg, 256, 0, stream>>>(xb, wkb, bk, kraw, S_, DIM_, DIM_, 0);
  gemm_bf16_wmma<<<gg, 256, 0, stream>>>(xb, wvb, bv, vTb,  S_, DIM_, DIM_, 2); // transposed bf16

  // 3) rmsnorm + rope -> bf16 Q/K
  rmsnorm_rope_kernel<<<S_, 256, 0, stream>>>(qraw, gq, freqs, gsz, qbuf);
  rmsnorm_rope_kernel<<<S_, 256, 0, stream>>>(kraw, gk, freqs, gsz, kbuf);

  // 4) flash attention (WMMA + TDM-staged K/V tiles)
  attention_kernel<<<NH_ * (S_ / 64), 128, 0, stream>>>(qbuf, kbuf, vTb, seqlen, attn);

  // 5) output projection -> fp32 d_out
  gemm_bf16_wmma<<<gg, 256, 0, stream>>>(attn, wob, bo, (float*)d_out, S_, DIM_, DIM_, 0);
}
